// SparseDiffusionModel_85813446574850
// MI455X (gfx1250) — compile-verified, matrix operand order checked by reading, not run on hardware
//
#include <hip/hip_runtime.h>
#include <hip/hip_bf16.h>
#include <stdint.h>

typedef __attribute__((ext_vector_type(16))) _Float16 v16h;
typedef __attribute__((ext_vector_type(8)))  float    v8f;

#define N_TOT 131072
#define BSZ   64

// ---------------------------------------------------------------------------
// Async 32KB global->LDS copy (256 threads, 16B/lane/iter), ASYNCcnt-tracked.
// dsaddr uses low 32 bits of the generic shared pointer (= wave-relative LDS
// address per ISA flat-aperture mapping). GVS mode: SGPR base + VGPR offset.
// ---------------------------------------------------------------------------
__device__ __forceinline__ void async_copy_32k(const _Float16* __restrict__ gsrc,
                                               _Float16* __restrict__ sdst, int t) {
    uint64_t sbase = (uint64_t)(size_t)gsrc;
    uint32_t lds   = (uint32_t)(size_t)(void*)sdst + (uint32_t)t * 16u;
    uint32_t voff  = (uint32_t)t * 16u;
    #pragma unroll
    for (int it = 0; it < 8; ++it) {
        asm volatile("global_load_async_to_lds_b128 %0, %1, %2"
                     :: "v"(lds + (uint32_t)it * 4096u),
                        "v"(voff + (uint32_t)it * 4096u),
                        "s"(sbase)
                     : "memory");
    }
}
__device__ __forceinline__ void wait_async0() {
    asm volatile("s_wait_asynccnt 0" ::: "memory");
}

// ---------------------------------------------------------------------------
// 1) segment starts from sorted batch_idx (binary search per batch)
// ---------------------------------------------------------------------------
__global__ void seg_kernel(const int* __restrict__ bidx, int N, int* __restrict__ starts) {
    int b = threadIdx.x;           // 64 threads
    int lo = 0, hi = N;
    while (lo < hi) { int mid = (lo + hi) >> 1; if (bidx[mid] < b) lo = mid + 1; else hi = mid; }
    starts[b] = lo;
    if (b == 0) starts[BSZ] = N;
}

// ---------------------------------------------------------------------------
// 2a) Wtmp_{k,v} = Wtok @ W{k,v}   (fp32), bvec_{k,v} = btok @ W{k,v} + b{k,v}
// ---------------------------------------------------------------------------
__global__ void w1_kernel(const float* __restrict__ Wtok, const float* __restrict__ btok,
                          const float* __restrict__ Wk, const float* __restrict__ bk,
                          const float* __restrict__ Wv, const float* __restrict__ bv,
                          float* __restrict__ Wtmp, float* __restrict__ bvec) {
    int i = blockIdx.x;            // 128 blocks
    int t = threadIdx.x;           // 256 threads
    int w = t >> 7, c = t & 127;
    const float* W2 = w ? Wv : Wk;
    float a = 0.f;
    for (int j = 0; j < 128; ++j) a += Wtok[i * 128 + j] * W2[j * 128 + c];
    Wtmp[w * 16384 + i * 128 + c] = a;
    if (i == 0) {
        const float* bb = w ? bv : bk;
        float s = bb[c];
        for (int j = 0; j < 128; ++j) s += btok[j] * W2[j * 128 + c];
        bvec[w * 128 + c] = s;
    }
}

// ---------------------------------------------------------------------------
// 2b) W_eff = Wtmp @ Wi{k,v}, written PRE-SWIZZLED in WMMA B-fragment order:
//     half index = w*16384 + ((ct*4+kk)*32 + lane)*16 + i
//     lane: n = ct*16 + (lane&15), kb = (lane>>4)*8
//     i:    K = kk*32 + (i<8 ? kb+i : 8+kb+i)
//     Also effbias = bvec @ Wi + bi.
// ---------------------------------------------------------------------------
__global__ void w2_kernel(const float* __restrict__ Wtmp, const float* __restrict__ bvec,
                          const float* __restrict__ Wik, const float* __restrict__ bik,
                          const float* __restrict__ Wiv, const float* __restrict__ biv,
                          _Float16* __restrict__ Weff, float* __restrict__ effbias) {
    int tid = blockIdx.x * 256 + threadIdx.x;   // 128 blocks * 256 = 32768
    int w  = tid >> 14;
    int r  = tid & 16383;
    int i  = r & 15;
    int L  = (r >> 4) & 31;
    int fg = r >> 9;                // ct*4 + kk
    int kk = fg & 3, ct = fg >> 2;
    int n  = ct * 16 + (L & 15);
    int kb = (L >> 4) * 8;
    int K  = kk * 32 + ((i < 8) ? (kb + i) : (8 + kb + i));
    const float* Wi = w ? Wiv : Wik;
    const float* A  = Wtmp + w * 16384 + K * 128;
    float a = 0.f;
    for (int j = 0; j < 128; ++j) a += A[j] * Wi[j * 128 + n];
    Weff[tid] = (_Float16)a;
    if (blockIdx.x == 0) {
        int t = threadIdx.x; int ww = t >> 7; int c = t & 127;
        const float* Wi2 = ww ? Wiv : Wik;
        const float* bi  = ww ? biv : bik;
        float s = bi[c];
        for (int j = 0; j < 128; ++j) s += bvec[ww * 128 + j] * Wi2[j * 128 + c];
        effbias[ww * 128 + c] = s;
    }
}

// ---------------------------------------------------------------------------
// 3) per-batch query path: timestep embedding -> time MLP -> +input proj
//    -> q0 = y@Wq+bq -> q = q0@Wiq+biq     (one 128-thread block per batch)
// ---------------------------------------------------------------------------
__global__ void q_kernel(const float* __restrict__ y_t, const int* __restrict__ tarr,
                         const float* __restrict__ Wt, const float* __restrict__ bt,
                         const float* __restrict__ Win, const float* __restrict__ bin_,
                         const float* __restrict__ Wq, const float* __restrict__ bq,
                         const float* __restrict__ Wiq, const float* __restrict__ biq,
                         float* __restrict__ qg) {
    int b = blockIdx.x, t = threadIdx.x;       // 128 threads
    __shared__ float emb[64], ys[512], q0s[128];
    if (t < 32) {
        float fr = __expf(-(float)t * 0.28782313662425574f);   // ln(10000)/32
        float a  = (float)tarr[b] * fr;
        emb[t]      = __sinf(a);
        emb[t + 32] = __cosf(a);
    }
    __syncthreads();
    for (int hh = t; hh < 512; hh += 128) {
        float accT = bt[hh];
        for (int j = 0; j < 64; ++j) accT += emb[j] * Wt[j * 512 + hh];
        float accI = bin_[hh];
        for (int j = 0; j < 3; ++j) accI += y_t[b * 3 + j] * Win[j * 512 + hh];
        ys[hh] = fmaxf(accT, 0.f) + accI;
    }
    __syncthreads();
    {
        float a = bq[t];
        for (int j = 0; j < 512; ++j) a += ys[j] * Wq[j * 128 + t];
        q0s[t] = a;
    }
    __syncthreads();
    {
        float a = biq[t];
        for (int j = 0; j < 128; ++j) a += q0s[j] * Wiq[j * 128 + t];
        qg[b * 128 + t] = a;
    }
}

// ---------------------------------------------------------------------------
// 4) fused K/V GEMM + scores, WMMA. Per wave: 16x128 strip.
//    Phase K: async-load Wk_eff to LDS, WMMA K-tiles -> K strip stays in LDS,
//             scores = q . k computed in-kernel (K never written to HBM).
//    Phase V: Wv_eff async copy overlapped with score phase; WMMA V -> HBM f16.
// ---------------------------------------------------------------------------
__global__ void __launch_bounds__(256) kv_kernel(const float* __restrict__ feat,
                                                 const _Float16* __restrict__ Weff,
                                                 const float* __restrict__ effbias,
                                                 const int* __restrict__ bidx,
                                                 const float* __restrict__ qg,
                                                 _Float16* __restrict__ vbuf,
                                                 float* __restrict__ scores) {
    __shared__ __align__(32) _Float16 sW[16384];       // 32 KB: one fused weight matrix
    __shared__ __align__(32) _Float16 sK[8 * 2048];    // 32 KB: per-wave 16x128 K strip
    int t = threadIdx.x;

    async_copy_32k(Weff, sW, t);                       // Wk_eff -> LDS (ASYNCcnt)

    int wave = t >> 5, lane = t & 31;
    int row_base = blockIdx.x * 128 + wave * 16;
    int m = lane & 15, kb = (lane >> 4) * 8;
    int mofs = (lane >> 4) * 8;
    const float* arow = feat + (size_t)(row_base + m) * 128;
    __builtin_prefetch(arow, 0, 3);

    // A fragments: 16x32 f16 per K-step, ISA layout (K = kk*32 + kb + {0..7, 16..23})
    v16h a[4];
    #pragma unroll
    for (int kk = 0; kk < 4; ++kk) {
        int K0 = kk * 32 + kb;
        #pragma unroll
        for (int i2 = 0; i2 < 8; ++i2) {
            a[kk][i2]     = (_Float16)arow[K0 + i2];
            a[kk][i2 + 8] = (_Float16)arow[K0 + 16 + i2];
        }
    }

    wait_async0();
    __syncthreads();

    // ---- K phase: 32 WMMAs, results parked in LDS
    _Float16* sKw = sK + wave * 2048;
    #pragma unroll
    for (int ct = 0; ct < 8; ++ct) {
        float bias = effbias[ct * 16 + m];
        v8f acc;
        #pragma unroll
        for (int r = 0; r < 8; ++r) acc[r] = bias;
        #pragma unroll
        for (int kk = 0; kk < 4; ++kk) {
            const v16h bfrag = *(const v16h*)&sW[((ct * 4 + kk) * 32 + lane) * 16];
            acc = __builtin_amdgcn_wmma_f32_16x16x32_f16(
                    false, a[kk], false, bfrag, (short)0, acc, false, false);
        }
        #pragma unroll
        for (int r = 0; r < 8; ++r)
            sKw[(mofs + r) * 128 + ct * 16 + m] = (_Float16)acc[r];
    }
    __syncthreads();                                   // K strips visible; sW reads done

    async_copy_32k(Weff + 16384, sW, t);               // Wv_eff -> LDS, overlapped below

    // ---- score phase: each lane -> (row = lane&15, heads {h0, h0+1})
    {
        int r_loc = lane & 15;
        int h0 = (lane >> 4) * 2;
        int n = row_base + r_loc;
        int bb = bidx[n];
        const float* qrow = qg + (size_t)bb * 128;
        const _Float16* krow = sKw + r_loc * 128;
        #pragma unroll
        for (int hh = 0; hh < 2; ++hh) {
            int h = h0 + hh;
            float acc2 = 0.f;
            #pragma unroll
            for (int d = 0; d < 32; ++d)
                acc2 += (float)krow[h * 32 + d] * qrow[h * 32 + d];
            scores[(size_t)n * 4 + h] = acc2 * 0.17677669529663687f;   // 1/sqrt(32)
        }
    }

    wait_async0();
    __syncthreads();

    // ---- V phase: 32 WMMAs, stored f16 to HBM
    #pragma unroll
    for (int ct = 0; ct < 8; ++ct) {
        float bias = effbias[128 + ct * 16 + m];
        v8f acc;
        #pragma unroll
        for (int r = 0; r < 8; ++r) acc[r] = bias;
        #pragma unroll
        for (int kk = 0; kk < 4; ++kk) {
            const v16h bfrag = *(const v16h*)&sW[((ct * 4 + kk) * 32 + lane) * 16];
            acc = __builtin_amdgcn_wmma_f32_16x16x32_f16(
                    false, a[kk], false, bfrag, (short)0, acc, false, false);
        }
        #pragma unroll
        for (int r = 0; r < 8; ++r)
            vbuf[(size_t)(row_base + mofs + r) * 128 + ct * 16 + m] = (_Float16)acc[r];
    }
}

// ---------------------------------------------------------------------------
// 5) per-batch segment softmax + V reduction. Scores precomputed by kv_kernel.
// ---------------------------------------------------------------------------
__global__ void __launch_bounds__(256) att_kernel(const _Float16* __restrict__ vbuf,
                                                  const float* __restrict__ scores,
                                                  const int* __restrict__ starts,
                                                  float* __restrict__ ctx_out) {
    int b = blockIdx.x, t = threadIdx.x;   // 256 threads
    __shared__ float red[256];
    __shared__ float wbuf[64 * 4];
    __shared__ float m_sh[4], d_sh[4];
    __shared__ float accbuf[2][128];
    int s = starts[b], e = starts[b + 1];

    // pass 1: streaming max over precomputed scores (float4 per row)
    const float4* sc4 = (const float4*)scores;
    float mx[4] = {-1e30f, -1e30f, -1e30f, -1e30f};
    for (int n = s + t; n < e; n += 256) {
        float4 v = sc4[n];
        mx[0] = fmaxf(mx[0], v.x); mx[1] = fmaxf(mx[1], v.y);
        mx[2] = fmaxf(mx[2], v.z); mx[3] = fmaxf(mx[3], v.w);
    }
    for (int h = 0; h < 4; ++h) {
        red[t] = mx[h]; __syncthreads();
        for (int off = 128; off > 0; off >>= 1) {
            if (t < off) red[t] = fmaxf(red[t], red[t + off]);
            __syncthreads();
        }
        if (t == 0) m_sh[h] = red[0];
        __syncthreads();
    }

    // pass 2: chunk 64 rows; stage softmax weights then accumulate V
    float acc = 0.f, dpart = 0.f;
    int h_mine = t & 3, nl_mine = t >> 2;
    int g = t >> 7, d = t & 127, hd = d >> 5;
    for (int base = s; base < e; base += 64) {
        int n = base + nl_mine;
        float w = 0.f;
        if (n < e) w = __expf(scores[(size_t)n * 4 + h_mine] - m_sh[h_mine]);
        wbuf[nl_mine * 4 + h_mine] = w;
        dpart += w;
        __syncthreads();
        int lim = min(64, e - base);
        int n0 = g * 32, n1 = min(n0 + 32, lim);
        for (int nl = n0; nl < n1; ++nl)
            acc += wbuf[nl * 4 + hd] * (float)vbuf[(size_t)(base + nl) * 128 + d];
        __syncthreads();
    }
    red[t] = dpart; __syncthreads();
    if (t < 4) { float sd = 0.f; for (int i = t; i < 256; i += 4) sd += red[i]; d_sh[t] = sd; }
    accbuf[g][d] = acc;
    __syncthreads();
    if (t < 128) {
        float v = accbuf[0][t] + accbuf[1][t];
        ctx_out[b * 128 + t] = v / d_sh[t >> 5];
    }
}

// ---------------------------------------------------------------------------
// 6) final chain: ctx -> Wo -> Wco -> Wout (one 512-thread block per batch)
// ---------------------------------------------------------------------------
__global__ void out_kernel(const float* __restrict__ ctx_out,
                           const float* __restrict__ Wo, const float* __restrict__ bo,
                           const float* __restrict__ Wco, const float* __restrict__ bco,
                           const float* __restrict__ Wout, const float* __restrict__ bout,
                           float* __restrict__ out) {
    int b = blockIdx.x, t = threadIdx.x;   // 512 threads
    __shared__ float ctx[128], attn[128], y2[512];
    if (t < 128) ctx[t] = ctx_out[b * 128 + t];
    __syncthreads();
    if (t < 128) {
        float a = bo[t];
        for (int j = 0; j < 128; ++j) a += ctx[j] * Wo[j * 128 + t];
        attn[t] = a;
    }
    __syncthreads();
    {
        float a = bco[t];
        for (int j = 0; j < 128; ++j) a += attn[j] * Wco[j * 512 + t];
        y2[t] = a;
    }
    __syncthreads();
    if (t < 3) {
        float a = bout[t];
        for (int j = 0; j < 512; ++j) a += y2[j] * Wout[j * 3 + t];
        out[b * 3 + t] = a;
    }
}

// ---------------------------------------------------------------------------
extern "C" void kernel_launch(void* const* d_in, const int* in_sizes, int n_in,
                              void* d_out, int out_size, void* d_ws, size_t ws_size,
                              hipStream_t stream) {
    const float* features = (const float*)d_in[0];
    const int*   batch_idx= (const int*)d_in[1];
    const float* y_t      = (const float*)d_in[2];
    const int*   t_arr    = (const int*)d_in[3];
    // d_in[4] = N_max (unused: padding never affects the output)
    const float* Wtok = (const float*)d_in[5];  const float* btok = (const float*)d_in[6];
    const float* Wt   = (const float*)d_in[7];  const float* bt   = (const float*)d_in[8];
    const float* Win  = (const float*)d_in[9];  const float* bin_ = (const float*)d_in[10];
    const float* Wq   = (const float*)d_in[11]; const float* bq   = (const float*)d_in[12];
    const float* Wk   = (const float*)d_in[13]; const float* bk   = (const float*)d_in[14];
    const float* Wv   = (const float*)d_in[15]; const float* bv   = (const float*)d_in[16];
    const float* Wiq  = (const float*)d_in[17]; const float* biq  = (const float*)d_in[18];
    const float* Wik  = (const float*)d_in[19]; const float* bik  = (const float*)d_in[20];
    const float* Wiv  = (const float*)d_in[21]; const float* biv  = (const float*)d_in[22];
    const float* Wo   = (const float*)d_in[23]; const float* bo   = (const float*)d_in[24];
    const float* Wco  = (const float*)d_in[25]; const float* bco  = (const float*)d_in[26];
    const float* Wout = (const float*)d_in[27]; const float* bout = (const float*)d_in[28];
    float* out = (float*)d_out;

    const int N = in_sizes[1];                 // 131072

    // workspace layout (bytes)
    char* wsb = (char*)d_ws;
    float*    Wtmp    = (float*)(wsb + 0);                         // 131072
    float*    bvec    = (float*)(wsb + 131072);                    // 1024
    float*    effbias = (float*)(wsb + 132096);                    // 1024
    _Float16* Weff    = (_Float16*)(wsb + 133120);                 // 65536 (pre-swizzled)
    size_t o = 198656;
    _Float16* vbuf    = (_Float16*)(wsb + o); o += (size_t)N * 128 * 2;  // 33.5 MB
    float*    scores  = (float*)(wsb + o);    o += (size_t)N * 4 * 4;    // 2 MB
    float*    qg      = (float*)(wsb + o);    o += (size_t)BSZ * 128 * 4;
    int*      starts  = (int*)(wsb + o);      o += 512;
    float*    ctx     = (float*)(wsb + o);    o += (size_t)BSZ * 128 * 4;

    seg_kernel<<<1, 64, 0, stream>>>(batch_idx, N, starts);
    w1_kernel<<<128, 256, 0, stream>>>(Wtok, btok, Wk, bk, Wv, bv, Wtmp, bvec);
    w2_kernel<<<128, 256, 0, stream>>>(Wtmp, bvec, Wik, bik, Wiv, biv, Weff, effbias);
    q_kernel<<<BSZ, 128, 0, stream>>>(y_t, t_arr, Wt, bt, Win, bin_, Wq, bq, Wiq, biq, qg);
    kv_kernel<<<N / 128, 256, 0, stream>>>(features, Weff, effbias, batch_idx, qg, vbuf, scores);
    att_kernel<<<BSZ, 256, 0, stream>>>(vbuf, scores, starts, ctx);
    out_kernel<<<BSZ, 512, 0, stream>>>(ctx, Wo, bo, Wco, bco, Wout, bout, out);
}
